// Gate_73787538145968
// MI455X (gfx1250) — matrix operand order repaired and checked by reading, not compiled, not run on hardware
//
#include <hip/hip_runtime.h>
#include <math.h>

// MoE gate (DeepSeek-V3 style): sigmoid(x @ W^T), group-limited top-8 routing.
// x: [8192, 7168] f32, W: [256, 7168] f32, bias: [256] f32
// out: weights [8192,8] f32  ++  indices [8192,8] i32 (flat-concatenated)
//
// GEMM on v_wmma_f32_16x16x32_bf16 with split-bf16 decomposition:
//   x ~= xh + xl, W ~= wh + wl  (bf16 RNE), logits ~= xh*wh + xh*wl + xl*wh
// Per-wave tile: 32 tokens x 32 experts (4 accumulators) -> 24 WMMA per
// K=64 chunk vs 32 fragment loads; x staging is register+LDS double-buffered.

typedef __attribute__((ext_vector_type(8)))  float  v8f;
typedef __attribute__((ext_vector_type(8)))  __bf16 v8bf;
typedef __attribute__((ext_vector_type(16))) __bf16 v16bf;

#define DIM 7168
#define N_EXPERTS 256
#define TPB 32           // tokens per block
#define BK 64            // K-chunk (two K=32 WMMA steps)
#define XSTR 72          // LDS row stride (bf16 elems): 144B rows
#define XBUF (TPB * XSTR)
#define SSTR 258         // LDS score row stride (f32, padded)
#define TOPK 8
#define NGROUPS 8
#define TOPK_GROUPS 4
#define ROUTE_SCALE 2.5f

// ---- bf16 split helpers (round-to-nearest-even) ----
__device__ __forceinline__ unsigned int bf16_rne(float f) {
    unsigned int u = __float_as_uint(f);
    return (u + 0x7fffu + ((u >> 16) & 1u)) >> 16;
}
__device__ __forceinline__ void split_bf16(float f, unsigned int& h, unsigned int& l) {
    h = bf16_rne(f);
    float hf = __uint_as_float(h << 16);
    l = bf16_rne(f - hf);
}

// A fragment (16x32 bf16 layout): two contiguous 8-elem K-runs from LDS.
__device__ __forceinline__ v16bf afrag(const unsigned short* row, int ko0, int ko1) {
    v8bf lo = *(const v8bf*)(const void*)(row + ko0);
    v8bf hi = *(const v8bf*)(const void*)(row + ko1);
    return __builtin_shufflevector(lo, hi, 0,1,2,3,4,5,6,7,8,9,10,11,12,13,14,15);
}

#define WMMA_BF16(C, A, B) \
    C = __builtin_amdgcn_wmma_f32_16x16x32_bf16(false, A, false, B, (short)0, C, false, false)

// ---- one-shot W -> (wh, wl) bf16 split ----
__global__ __launch_bounds__(256)
void convert_w_kernel(const float* __restrict__ w,
                      unsigned short* __restrict__ wh,
                      unsigned short* __restrict__ wl, int npairs)
{
    int i = blockIdx.x * blockDim.x + threadIdx.x;
    if (i >= npairs) return;
    float2 f = ((const float2*)w)[i];
    unsigned int h0, l0, h1, l1;
    split_bf16(f.x, h0, l0);
    split_bf16(f.y, h1, l1);
    *(unsigned int*)&wh[2 * i] = h0 | (h1 << 16);
    *(unsigned int*)&wl[2 * i] = l0 | (l1 << 16);
}

__global__ __launch_bounds__(256, 1)
void moe_gate_kernel(const float* __restrict__ x,
                     const unsigned short* __restrict__ whb,
                     const unsigned short* __restrict__ wlb,
                     const float* __restrict__ bias,
                     float* __restrict__ wout,
                     int* __restrict__ iout)
{
    __shared__ unsigned short XH[2 * XBUF];     // x-hi tiles (double buffered)
    __shared__ unsigned short XL[2 * XBUF];     // x-lo tiles
    __shared__ float Scores[TPB * SSTR];        // 32 x 256 sigmoid scores

    const int tid  = threadIdx.x;
    const int wave = tid >> 5;        // 0..7
    const int lane = tid & 31;
    const int half = lane >> 4;       // lane half selects K sub-runs
    const int l16  = lane & 15;
    const int t0   = blockIdx.x * TPB;

    // B operands: this lane's two expert rows (wave covers 32 experts)
    const int e0 = wave * 32 + l16;
    const int e1 = e0 + 16;
    const unsigned short* wh0 = whb + (size_t)e0 * DIM;
    const unsigned short* wl0 = wlb + (size_t)e0 * DIM;
    const unsigned short* wh1 = whb + (size_t)e1 * DIM;
    const unsigned short* wl1 = wlb + (size_t)e1 * DIM;

    // cooperative x staging: thread -> (row, 8 cols) of the 32x64 f32 tile
    const int srow = tid >> 3;            // 0..31
    const int scol = (tid & 7) * 8;       // 0..56
    const float* xsrc = x + (size_t)(t0 + srow) * DIM + scol;

    v8f c00 = {}, c01 = {};   // tokens [0..15]  x experts e0/e1 tiles
    v8f c10 = {}, c11 = {};   // tokens [16..31] x experts e0/e1 tiles

    // register double-buffer for the x tile
    float4 curA = *(const float4*)(xsrc);
    float4 curB = *(const float4*)(xsrc + 4);

    for (int kk = 0; kk < DIM; kk += BK) {
        const int base = ((kk >> 6) & 1) * XBUF;

        // convert 8 f32 -> packed bf16 hi/lo, one b128 store per array
        unsigned int h[8], l[8];
        split_bf16(curA.x, h[0], l[0]); split_bf16(curA.y, h[1], l[1]);
        split_bf16(curA.z, h[2], l[2]); split_bf16(curA.w, h[3], l[3]);
        split_bf16(curB.x, h[4], l[4]); split_bf16(curB.y, h[5], l[5]);
        split_bf16(curB.z, h[6], l[6]); split_bf16(curB.w, h[7], l[7]);
        uint4 hp = make_uint4(h[0] | (h[1] << 16), h[2] | (h[3] << 16),
                              h[4] | (h[5] << 16), h[6] | (h[7] << 16));
        uint4 lp = make_uint4(l[0] | (l[1] << 16), l[2] | (l[3] << 16),
                              l[4] | (l[5] << 16), l[6] | (l[7] << 16));
        *(uint4*)&XH[base + srow * XSTR + scol] = hp;
        *(uint4*)&XL[base + srow * XSTR + scol] = lp;

        // fetch next x chunk into registers (overlaps compute below)
        if (kk + BK < DIM) {
            curA = *(const float4*)(xsrc + kk + BK);
            curB = *(const float4*)(xsrc + kk + BK + 4);
            __builtin_prefetch(wh0 + kk + BK, 0, 1);
            __builtin_prefetch(wl0 + kk + BK, 0, 1);
            __builtin_prefetch(wh1 + kk + BK, 0, 1);
            __builtin_prefetch(wl1 + kk + BK, 0, 1);
        }
        __syncthreads();

#pragma unroll
        for (int s = 0; s < 2; ++s) {           // two K=32 steps
            const int kb = kk + s * 32;
            // B frags 32x16 bf16: one contiguous 16-K run per lane
            v16bf bh0 = *(const v16bf*)(const void*)(wh0 + kb + half * 16);
            v16bf bl0 = *(const v16bf*)(const void*)(wl0 + kb + half * 16);
            v16bf bh1 = *(const v16bf*)(const void*)(wh1 + kb + half * 16);
            v16bf bl1 = *(const v16bf*)(const void*)(wl1 + kb + half * 16);
            // A frags 16x32 bf16: two 8-K runs at h*16 and h*16+8
            const int ko0 = s * 32 + half * 16;
            const int ko1 = s * 32 + half * 16 + 8;
            v16bf ah0 = afrag(&XH[base + l16 * XSTR], ko0, ko1);
            v16bf al0 = afrag(&XL[base + l16 * XSTR], ko0, ko1);
            v16bf ah1 = afrag(&XH[base + (l16 + 16) * XSTR], ko0, ko1);
            v16bf al1 = afrag(&XL[base + (l16 + 16) * XSTR], ko0, ko1);

            WMMA_BF16(c00, ah0, bh0);  WMMA_BF16(c01, ah0, bh1);
            WMMA_BF16(c10, ah1, bh0);  WMMA_BF16(c11, ah1, bh1);
            WMMA_BF16(c00, ah0, bl0);  WMMA_BF16(c01, ah0, bl1);
            WMMA_BF16(c10, ah1, bl0);  WMMA_BF16(c11, ah1, bl1);
            WMMA_BF16(c00, al0, bh0);  WMMA_BF16(c01, al0, bh1);
            WMMA_BF16(c10, al1, bh0);  WMMA_BF16(c11, al1, bh1);
        }
    }

    // ---- sigmoid + scatter into Scores[token][expert] ----
#pragma unroll
    for (int r = 0; r < 8; ++r) {
        int trow = r + half * 8;   // C/D: vgpr r holds M=r (lanes 0-15) / r+8
        Scores[trow * SSTR + e0]        = 1.0f / (1.0f + __expf(-c00[r]));
        Scores[trow * SSTR + e1]        = 1.0f / (1.0f + __expf(-c01[r]));
        Scores[(trow + 16) * SSTR + e0] = 1.0f / (1.0f + __expf(-c10[r]));
        Scores[(trow + 16) * SSTR + e1] = 1.0f / (1.0f + __expf(-c11[r]));
    }
    __syncthreads();

    // ---- routing: wave w routes tokens w, w+8, w+16, w+24 ----
    for (int tt = 0; tt < 4; ++tt) {
        const int token = wave + tt * 8;
        const float* srowp = &Scores[token * SSTR];

        float sb[8];
#pragma unroll
        for (int j = 0; j < 8; ++j) {
            int e = lane * 8 + j;
            sb[j] = srowp[e] + bias[e];
        }

        // lane-local top-2 of biased scores
        float m1 = -INFINITY, m2 = -INFINITY;
#pragma unroll
        for (int j = 0; j < 8; ++j) {
            float v = sb[j];
            if (v > m1)      { m2 = m1; m1 = v; }
            else if (v > m2) { m2 = v; }
        }
        // merge top-2 across the 4 lanes of this group (group = lane>>2)
#pragma unroll
        for (int d = 1; d <= 2; d <<= 1) {
            float o1 = __shfl_xor(m1, d, 32);
            float o2 = __shfl_xor(m2, d, 32);
            float n1 = fmaxf(m1, o1);
            float n2 = fmaxf(fminf(m1, o1), fmaxf(m2, o2));
            m1 = n1; m2 = n2;
        }
        float gscore = m1 + m2;

        // gather the 8 group scores to every lane
        float gsc[8];
#pragma unroll
        for (int g = 0; g < 8; ++g)
            gsc[g] = __shfl(gscore, g * 4, 32);

        // top-4 groups (ties -> lowest index, matching jax top_k)
        unsigned keepMask = 0;
#pragma unroll
        for (int r = 0; r < TOPK_GROUPS; ++r) {
            int bg = 0; float bv = -INFINITY;
#pragma unroll
            for (int g = 0; g < NGROUPS; ++g) {
                bool free = !((keepMask >> g) & 1u);
                if (free && gsc[g] > bv) { bv = gsc[g]; bg = g; }
            }
            keepMask |= (1u << bg);
        }
        if (!((keepMask >> (lane >> 2)) & 1u)) {
#pragma unroll
            for (int j = 0; j < 8; ++j) sb[j] = -INFINITY;
        }

        // global top-8: 8 rounds of wave-wide argmax (tie -> lowest index)
        float topw[8];
        int   topi[8];
        unsigned taken = 0;
        float wsum = 0.0f;
#pragma unroll
        for (int r = 0; r < TOPK; ++r) {
            float bv = -INFINITY; int be = N_EXPERTS;
#pragma unroll
            for (int j = 0; j < 8; ++j) {
                bool free = !((taken >> j) & 1u);
                if (free && sb[j] > bv) { bv = sb[j]; be = lane * 8 + j; }
            }
#pragma unroll
            for (int d = 1; d < 32; d <<= 1) {
                float ov = __shfl_xor(bv, d, 32);
                int   oe = __shfl_xor(be, d, 32);
                if (ov > bv || (ov == bv && oe < be)) { bv = ov; be = oe; }
            }
            if ((be >> 3) == lane) taken |= 1u << (be & 7);
            float worig = srowp[be];   // original sigmoid score (LDS broadcast)
            topw[r] = worig;
            topi[r] = be;
            wsum += worig;
        }

        const float scale = ROUTE_SCALE / wsum;
        if (lane == 0) {
            float* wo = wout + (size_t)(t0 + token) * TOPK;
            int*   io = iout + (size_t)(t0 + token) * TOPK;
#pragma unroll
            for (int r = 0; r < TOPK; ++r) {
                wo[r] = topw[r] * scale;
                io[r] = topi[r];
            }
        }
    }
}

extern "C" void kernel_launch(void* const* d_in, const int* in_sizes, int n_in,
                              void* d_out, int out_size, void* d_ws, size_t ws_size,
                              hipStream_t stream) {
    const float* x    = (const float*)d_in[0];   // [8192, 7168]
    const float* w    = (const float*)d_in[1];   // [256, 7168]
    const float* bias = (const float*)d_in[2];   // [256]

    const int n_tokens = in_sizes[0] / DIM;      // 8192
    const int welems   = in_sizes[1];            // 256*7168

    // workspace: bf16 hi/lo copies of W (7.34 MB total)
    unsigned short* wh = (unsigned short*)d_ws;
    unsigned short* wl = wh + (size_t)welems;

    float* wout = (float*)d_out;                           // [T, 8] f32
    int*   iout = (int*)(wout + (size_t)n_tokens * TOPK);  // [T, 8] i32

    const int npairs = welems / 2;
    convert_w_kernel<<<dim3((npairs + 255) / 256), dim3(256), 0, stream>>>(w, wh, wl, npairs);

    moe_gate_kernel<<<dim3(n_tokens / TPB), dim3(256), 0, stream>>>(x, wh, wl, bias, wout, iout);
}
